// RWKV6Attention_30743375905053
// MI455X (gfx1250) — matrix operand order repaired and checked by reading, not compiled, not run on hardware
//
#include <hip/hip_runtime.h>
#include <hip/hip_bf16.h>
#include <math.h>

typedef __bf16 bf16_t;
typedef __attribute__((ext_vector_type(8)))  __bf16 v8bf;
typedef __attribute__((ext_vector_type(16))) __bf16 v16bf;
typedef __attribute__((ext_vector_type(8)))  float  v8f;

// Problem dims (fixed by the reference)
constexpr int Bb   = 4;
constexpr int Lseq = 4096;
constexpr int HID  = 2048;
constexpr int NH   = 32;
constexpr int DK   = 1024;
constexpr int DV   = 2048;
constexpr int HQK  = 32;   // DK/NH
constexpr int HV   = 64;   // DV/NH
constexpr int RLO  = 32;
constexpr int GLO  = 64;
constexpr int M    = Bb * Lseq;   // 16384 tokens
constexpr int ZP   = 192;         // 5*RLO=160 padded to a multiple of 64

// CDNA5 async global->LDS path (device pass only; sync fallback otherwise)
#if defined(__gfx1250__) && \
    __has_builtin(__builtin_amdgcn_global_load_async_to_lds_b128) && \
    __has_builtin(__builtin_amdgcn_s_wait_asynccnt)
#define USE_ASYNC_LDS 1
#else
#define USE_ASYNC_LDS 0
#endif

#if USE_ASYNC_LDS
typedef int v4i_ __attribute__((__vector_size__(4 * sizeof(int))));
typedef __attribute__((address_space(1))) v4i_* gbl_v4i_p;   // global-space v4i*
typedef __attribute__((address_space(3))) v4i_* lds_v4i_p;   // LDS-space v4i*
#endif

__device__ inline void async_b128(bf16_t* lds, const bf16_t* g) {
#if USE_ASYNC_LDS
  __builtin_amdgcn_global_load_async_to_lds_b128(
      (gbl_v4i_p)g, (lds_v4i_p)lds, 0, 0);
#else
  *(uint4*)lds = *(const uint4*)g;
#endif
}
__device__ inline void wait_async() {
#if USE_ASYNC_LDS
  __builtin_amdgcn_s_wait_asynccnt(0);
#endif
}

// ---------------------------------------------------------------------------
// Fused fp32->bf16 convert + transpose: src row-major [K][N] fp32,
// dst row-major [Npad][K] bf16 (rows n>=N zeroed). K,Npad multiples of 32.
// Weights are tiny and reused by ~4k GEMM blocks -> transpose once so GEMM
// B-tiles are contiguous and can stream via async-to-LDS (no b16 scatter).
// ---------------------------------------------------------------------------
__global__ void __launch_bounds__(256)
cvt_transpose(bf16_t* __restrict__ dst, const float* __restrict__ src,
              int K, int N) {
  __shared__ float tile[32][33];
  const int k0 = blockIdx.x * 32, n0 = blockIdx.y * 32;
  for (int i = threadIdx.y; i < 32; i += 8) {
    int n = n0 + (int)threadIdx.x;
    tile[i][threadIdx.x] = (n < N) ? src[(size_t)(k0 + i) * N + n] : 0.0f;
  }
  __syncthreads();
  for (int i = threadIdx.y; i < 32; i += 8)
    dst[(size_t)(n0 + i) * K + k0 + threadIdx.x] = (bf16_t)tile[threadIdx.x][i];
}

// ---------------------------------------------------------------------------
// Token-shift prep: delta[t] = x[t-1]-x[t] (zero at t=0, per batch),
// xbar = bf16(x + delta*mu_x) (input of x_proj GEMM)
// ---------------------------------------------------------------------------
__global__ void prep_kernel(const float* __restrict__ x, const float* __restrict__ mu,
                            float* __restrict__ delta, bf16_t* __restrict__ xbar) {
  size_t i = (size_t)blockIdx.x * blockDim.x + threadIdx.x;
  if (i >= (size_t)M * HID) return;
  int    col = (int)(i % HID);
  size_t m   = i / HID;
  int    t   = (int)(m % Lseq);
  float xv = x[i];
  float xp = t ? x[i - HID] : 0.0f;
  float d  = xp - xv;
  delta[i] = d;
  xbar[i]  = (bf16_t)(xv + d * mu[col]);
}

// ---------------------------------------------------------------------------
// Generic bf16 WMMA GEMM: C = epi(A[M,K] @ W^T), W pre-transposed [N][K].
// Block tile 128x64, BK=32, 8 waves (4x2), each wave 32x32 (2x2 WMMA frags),
// double-buffered LDS, async global->LDS staging for both A and B tiles.
// ---------------------------------------------------------------------------
constexpr int BM = 128, BN = 64, BK = 32;
enum { EPI_F32 = 0, EPI_TANH_BF16 = 1, EPI_LERP_BF16 = 2, EPI_EXPEXP_F32 = 3 };

__device__ inline v16bf mk16(v8bf lo, v8bf hi) {
  v16bf r;
#pragma unroll
  for (int i = 0; i < 8; ++i) { r[i] = lo[i]; r[i + 8] = hi[i]; }
  return r;
}

template <int EPI>
__global__ void __launch_bounds__(256)
gemm_wmma(const bf16_t* __restrict__ A, int lda,
          const bf16_t* __restrict__ Bt, int ldb,   // transposed weights [N][K]
          void* __restrict__ Cv, int ldc, int K,
          const float* __restrict__ auxX,   // EPI_LERP: x (fp32, pitch ldc)
          const float* __restrict__ auxD,   // EPI_LERP: delta (fp32, pitch ldc)
          const float* __restrict__ bias)   // EPI_EXPEXP: bw[N]
{
  __shared__ __align__(16) bf16_t As[2][BM][BK];
  __shared__ __align__(16) bf16_t Bs[2][BN][BK];   // [n][k], contiguous K

  const int tid   = threadIdx.x;
  const int lane  = tid & 31;
  const int wid   = tid >> 5;
  const int wm    = wid & 3;    // wave tile along M (x32)
  const int wn    = wid >> 2;   // wave tile along N (x32)
  const int g     = lane >> 4;  // lane group (CDNA5 wave32 WMMA layout)
  const int rr    = lane & 15;
  const int mBase = blockIdx.x * BM;
  const int nBase = blockIdx.y * BN;

  // Loop-invariant staging coordinates: A tile = 512 x 16B chunks (2/thread),
  // B tile = 256 x 16B chunks (1/thread). Both contiguous along K.
  const int ar = tid >> 2;             // 0..63 (and +64)
  const int ac = (tid & 3) * 8;        // K offset within tile
  const bf16_t* aSrc0 = A  + (size_t)(mBase + ar)      * lda + ac;
  const bf16_t* aSrc1 = A  + (size_t)(mBase + ar + 64) * lda + ac;
  const bf16_t* bSrc  = Bt + (size_t)(nBase + ar)      * ldb + ac;

  auto stage = [&](int buf, int k0) {
    async_b128(&As[buf][ar][ac],      aSrc0 + k0);
    async_b128(&As[buf][ar + 64][ac], aSrc1 + k0);
    async_b128(&Bs[buf][ar][ac],      bSrc  + k0);
  };

  v8f acc[2][2] = {};
  const int nk = K / BK;
  stage(0, 0);
  wait_async();
  __syncthreads();

  for (int kt = 0; kt < nk; ++kt) {
    const int cur = kt & 1;
    if (kt + 1 < nk) stage(cur ^ 1, (kt + 1) * BK);  // overlap with WMMA below

    v16bf af[2], bfr[2];
#pragma unroll
    for (int i = 0; i < 2; ++i) {
      // A frag (16x32 bf16): lane(r,g) holds K = 8g+e (e<8), 16+8g+(e-8) (e>=8)
      int row = wm * 32 + i * 16 + rr;
      v8bf lo = *(const v8bf*)&As[cur][row][8 * g];
      v8bf hi = *(const v8bf*)&As[cur][row][16 + 8 * g];
      af[i] = mk16(lo, hi);
    }
#pragma unroll
    for (int j = 0; j < 2; ++j) {
      // B frag (32x16 bf16): lane(c,g) holds K = 8g+e (e<8), 16+8g+(e-8) (e>=8)
      int col = wn * 32 + j * 16 + rr;
      v8bf lo = *(const v8bf*)&Bs[cur][col][8 * g];
      v8bf hi = *(const v8bf*)&Bs[cur][col][16 + 8 * g];
      bfr[j] = mk16(lo, hi);
    }
#pragma unroll
    for (int i = 0; i < 2; ++i)
#pragma unroll
      for (int j = 0; j < 2; ++j)
        acc[i][j] = __builtin_amdgcn_wmma_f32_16x16x32_bf16(
            false, af[i], false, bfr[j], (short)0, acc[i][j], false, false);

    wait_async();     // next tile fully resident in LDS
    __syncthreads();
  }

  // Epilogue. C/D layout: VGPR v -> M = v + 8*g, N = rr (within 16x16 tile).
#pragma unroll
  for (int i = 0; i < 2; ++i) {
#pragma unroll
    for (int j = 0; j < 2; ++j) {
      const int rowb = mBase + wm * 32 + i * 16;
      const int n    = nBase + wn * 32 + j * 16 + rr;
#pragma unroll
      for (int vg = 0; vg < 8; ++vg) {
        const int    m   = rowb + vg + 8 * g;
        const float  val = acc[i][j][vg];
        const size_t off = (size_t)m * ldc + n;
        if constexpr (EPI == EPI_F32) {
          ((float*)Cv)[off] = val;
        } else if constexpr (EPI == EPI_TANH_BF16) {
          ((bf16_t*)Cv)[off] = (bf16_t)tanhf(val);
        } else if constexpr (EPI == EPI_LERP_BF16) {
          ((bf16_t*)Cv)[off] = (bf16_t)(auxX[off] + auxD[off] * val);
        } else {  // EPI_EXPEXP_F32: decay = exp(-exp(w + bw))
          ((float*)Cv)[off] = __expf(-__expf(val + bias[n]));
        }
      }
    }
  }
}

// ---------------------------------------------------------------------------
// RWKV6 sequential scan: one wave32 per (b,h). State S[32k x 64v] lives in
// registers: lane j owns v-columns j and j+32 (S0[i],S1[i], i=0..31).
// r/k/decay for step t broadcast through LDS (conflict-free same-addr reads).
//   kv = k(x)v ; o = r.(S + u*kv) ; S = S*decay + kv
// ---------------------------------------------------------------------------
__global__ void __launch_bounds__(32)
rwkv_scan(const float* __restrict__ r, const float* __restrict__ k,
          const float* __restrict__ v, const float* __restrict__ w,
          const float* __restrict__ bonus, float* __restrict__ o)
{
  const int bh   = blockIdx.x;
  const int b    = bh / NH;
  const int h    = bh % NH;
  const int lane = threadIdx.x;
  __shared__ float rs[HQK], ks[HQK], ws_[HQK], us[HQK];
  us[lane] = bonus[h * HQK + lane];
  float S0[HQK], S1[HQK];
#pragma unroll
  for (int i = 0; i < HQK; ++i) { S0[i] = 0.0f; S1[i] = 0.0f; }

  for (int t = 0; t < Lseq; ++t) {
    const size_t m  = (size_t)b * Lseq + t;
    const size_t bk = m * DK + (size_t)h * HQK;
    const size_t bv = m * DV + (size_t)h * HV;
    rs[lane]  = r[bk + lane];
    ks[lane]  = k[bk + lane];
    ws_[lane] = w[bk + lane];
    __syncthreads();  // single wave -> S_NOP; DS kept in order per wave
    const float v0 = v[bv + lane];
    const float v1 = v[bv + 32 + lane];
    float o0 = 0.0f, o1 = 0.0f;
#pragma unroll
    for (int i = 0; i < HQK; ++i) {
      const float ri = rs[i], ki = ks[i], di = ws_[i], ui = us[i];
      const float kv0 = ki * v0, kv1 = ki * v1;
      o0 += ri * (S0[i] + ui * kv0);
      o1 += ri * (S1[i] + ui * kv1);
      S0[i] = S0[i] * di + kv0;
      S1[i] = S1[i] * di + kv1;
    }
    o[bv + lane]      = o0;
    o[bv + 32 + lane] = o1;
    __syncthreads();
  }
}

// ---------------------------------------------------------------------------
// Fused per-head LayerNorm + SiLU gate, emit bf16 for the final WMMA GEMM.
// One thread per (token, head): 64 values in registers.
// ---------------------------------------------------------------------------
__global__ void gate_ln(const float* __restrict__ o, const float* __restrict__ g,
                        const float* __restrict__ gw, const float* __restrict__ gb,
                        bf16_t* __restrict__ ob)
{
  int idx = blockIdx.x * blockDim.x + threadIdx.x;   // over M*NH
  if (idx >= M * NH) return;
  const size_t base = (size_t)idx * HV;              // [M, NH*HV] contiguous
  float vals[HV];
  float mean = 0.0f, sq = 0.0f;
#pragma unroll 8
  for (int c = 0; c < HV; ++c) {
    float t = o[base + c];
    vals[c] = t; mean += t; sq += t * t;
  }
  mean *= (1.0f / HV);
  const float var = sq * (1.0f / HV) - mean * mean;
  const float inv = rsqrtf(var + 1e-5f);
#pragma unroll 8
  for (int c = 0; c < HV; ++c) {
    const float on = (vals[c] - mean) * inv * gw[c] + gb[c];
    const float gv = g[base + c];
    const float sw = gv / (1.0f + __expf(-gv));      // SiLU: g*sigmoid(g)
    ob[base + c] = (bf16_t)(on * sw);
  }
}

// ---------------------------------------------------------------------------
// Host orchestration
// ---------------------------------------------------------------------------
extern "C" void kernel_launch(void* const* d_in, const int* in_sizes, int n_in,
                              void* d_out, int out_size, void* d_ws, size_t ws_size,
                              hipStream_t stream) {
  (void)in_sizes; (void)n_in; (void)out_size; (void)ws_size;

  const float* x     = (const float*)d_in[0];
  const float* mu_x  = (const float*)d_in[1];
  const float* Wx1   = (const float*)d_in[2];
  const float* Wx2   = (const float*)d_in[3];
  const float* Wr    = (const float*)d_in[4];
  const float* Ww1   = (const float*)d_in[5];
  const float* Ww2   = (const float*)d_in[6];
  const float* bw    = (const float*)d_in[7];
  const float* Wk    = (const float*)d_in[8];
  const float* Wv    = (const float*)d_in[9];
  const float* Wg    = (const float*)d_in[10];
  const float* bonus = (const float*)d_in[11];
  const float* gnw   = (const float*)d_in[12];
  const float* gnb   = (const float*)d_in[13];
  const float* Wo    = (const float*)d_in[14];
  float* out = (float*)d_out;

  // --- workspace carve-out ---
  char* p = (char*)d_ws;
  auto alloc = [&](size_t bytes) -> void* {
    void* r = (void*)p;
    p += (bytes + 255) & ~(size_t)255;
    return r;
  };
  float*  delta = (float*)alloc((size_t)M * HID * 4);
  bf16_t* xbar  = (bf16_t*)alloc((size_t)M * HID * 2);
  bf16_t* zbuf  = (bf16_t*)alloc((size_t)M * ZP * 2);
  bf16_t* lerped[5];
  for (int i = 0; i < 5; ++i) lerped[i] = (bf16_t*)alloc((size_t)M * HID * 2);
  bf16_t* hlora = (bf16_t*)alloc((size_t)M * GLO * 2);
  float*  rbuf  = (float*)alloc((size_t)M * DK * 4);
  float*  kbuf  = (float*)alloc((size_t)M * DK * 4);
  float*  dbuf  = (float*)alloc((size_t)M * DK * 4);
  float*  vbuf  = (float*)alloc((size_t)M * DV * 4);
  float*  gbuf  = (float*)alloc((size_t)M * DV * 4);
  float*  obuf  = (float*)alloc((size_t)M * DV * 4);
  bf16_t* obf   = (bf16_t*)alloc((size_t)M * DV * 2);
  // transposed bf16 weights [N][K]
  bf16_t* Wx1t = (bf16_t*)alloc((size_t)ZP * HID * 2);
  bf16_t* Wx2t = (bf16_t*)alloc((size_t)5 * HID * RLO * 2);
  bf16_t* Wrt  = (bf16_t*)alloc((size_t)DK * HID * 2);
  bf16_t* Ww1t = (bf16_t*)alloc((size_t)GLO * HID * 2);
  bf16_t* Ww2t = (bf16_t*)alloc((size_t)DK * GLO * 2);
  bf16_t* Wkt  = (bf16_t*)alloc((size_t)DK * HID * 2);
  bf16_t* Wvt  = (bf16_t*)alloc((size_t)DV * HID * 2);
  bf16_t* Wgt  = (bf16_t*)alloc((size_t)DV * HID * 2);
  bf16_t* Wot  = (bf16_t*)alloc((size_t)HID * DV * 2);

  // Fused convert+transpose (dst [Npad][K] bf16), one-shot, reused by all tiles
  const dim3 tb(32, 8);
  auto ct = [&](bf16_t* dst, const float* src, int K, int N, int Npad) {
    cvt_transpose<<<dim3(K / 32, Npad / 32), tb, 0, stream>>>(dst, src, K, N);
  };
  ct(Wx1t, Wx1, HID, 5 * RLO, ZP);                       // [192][2048], rows 160+ zero
  for (int n = 0; n < 5; ++n)                            // per-branch [2048][32]
    ct(Wx2t + (size_t)n * HID * RLO, Wx2 + (size_t)n * RLO * HID, RLO, HID, HID);
  ct(Wrt,  Wr,  HID, DK, DK);
  ct(Ww1t, Ww1, HID, GLO, GLO);
  ct(Ww2t, Ww2, GLO, DK, DK);
  ct(Wkt,  Wk,  HID, DK, DK);
  ct(Wvt,  Wv,  HID, DV, DV);
  ct(Wgt,  Wg,  HID, DV, DV);
  ct(Wot,  Wo,  DV, HID, HID);

  // 1) token shift + x_proj lerp input
  prep_kernel<<<dim3(((size_t)M * HID + 255) / 256), dim3(256), 0, stream>>>(x, mu_x, delta, xbar);

  const dim3 blk(256);
  // 2) z = tanh(xbar @ Wx1)   [M,2048]x[2048,192] -> bf16
  gemm_wmma<EPI_TANH_BF16><<<dim3(M / BM, ZP / BN), blk, 0, stream>>>(
      xbar, HID, Wx1t, HID, zbuf, ZP, HID, nullptr, nullptr, nullptr);
  // 3) per-branch mu expansion fused with data-dependent lerp:
  //    lerped_n = bf16(x + delta * (z_n @ Wx2[n]))    K=32
  for (int n = 0; n < 5; ++n) {
    gemm_wmma<EPI_LERP_BF16><<<dim3(M / BM, HID / BN), blk, 0, stream>>>(
        zbuf + n * RLO, ZP, Wx2t + (size_t)n * HID * RLO, RLO,
        lerped[n], HID, RLO, x, delta, nullptr);
  }
  // 4) projections (branch order: r, w, k, v, g)
  gemm_wmma<EPI_F32><<<dim3(M / BM, DK / BN), blk, 0, stream>>>(
      lerped[0], HID, Wrt, HID, rbuf, DK, HID, nullptr, nullptr, nullptr);
  gemm_wmma<EPI_TANH_BF16><<<dim3(M / BM, GLO / BN), blk, 0, stream>>>(
      lerped[1], HID, Ww1t, HID, hlora, GLO, HID, nullptr, nullptr, nullptr);
  gemm_wmma<EPI_EXPEXP_F32><<<dim3(M / BM, DK / BN), blk, 0, stream>>>(
      hlora, GLO, Ww2t, GLO, dbuf, DK, GLO, nullptr, nullptr, bw);  // decay=exp(-exp(w+bw))
  gemm_wmma<EPI_F32><<<dim3(M / BM, DK / BN), blk, 0, stream>>>(
      lerped[2], HID, Wkt, HID, kbuf, DK, HID, nullptr, nullptr, nullptr);
  gemm_wmma<EPI_F32><<<dim3(M / BM, DV / BN), blk, 0, stream>>>(
      lerped[3], HID, Wvt, HID, vbuf, DV, HID, nullptr, nullptr, nullptr);
  gemm_wmma<EPI_F32><<<dim3(M / BM, DV / BN), blk, 0, stream>>>(
      lerped[4], HID, Wgt, HID, gbuf, DV, HID, nullptr, nullptr, nullptr);

  // 5) sequential RWKV scan: one wave32 per (b,h)
  rwkv_scan<<<dim3(Bb * NH), dim3(32), 0, stream>>>(rbuf, kbuf, vbuf, dbuf, bonus, obuf);

  // 6) LN + SiLU gate -> bf16
  gate_ln<<<dim3((M * NH + 255) / 256), dim3(256), 0, stream>>>(obuf, gbuf, gnw, gnb, obf);

  // 7) output projection -> d_out (fp32)
  gemm_wmma<EPI_F32><<<dim3(M / BM, HID / BN), blk, 0, stream>>>(
      obf, DV, Wot, DV, out, HID, DV, nullptr, nullptr, nullptr);
}
